// VectorQuantizerEMA_44169443672876
// MI455X (gfx1250) — matrix operand order repaired and checked by reading, not compiled, not run on hardware
//
#include <hip/hip_runtime.h>

// ---------------------------------------------------------------------------
// VQ-VAE eval forward for MI455X (gfx1250, wave32, WMMA 16x16x32 bf16).
// N=32768 queries, D=256, K=8192 codes.
// Distance GEMM via v_wmma_f32_16x16x32_bf16 with hi/lo bf16 split
// (x = hi + lo), dot ~= hi*hi' + hi*lo' + lo*hi'  (~2^-16 relative error).
// Code tiles staged LDS-double-buffered with GLOBAL_LOAD_ASYNC_TO_LDS_B128
// (ASYNCcnt) so the L2->LDS copy of tile t+1 overlaps the WMMAs of tile t.
// ---------------------------------------------------------------------------

typedef __attribute__((ext_vector_type(16))) __bf16 v16bf;
typedef __attribute__((ext_vector_type(8)))  float  v8f;

#define NQ    32768
#define DIM   256
#define KC    8192
#define QPB   128          // queries per block (8 waves * 16 rows)
#define NTILE (KC / 16)    // 512 code tiles

static __device__ __forceinline__ v8f wmma_bf16(v16bf a, v16bf b, v8f c) {
  // (neg_a, A, neg_b, B, c_mod, C, reuse_a, reuse_b)
  return __builtin_amdgcn_wmma_f32_16x16x32_bf16(false, a, false, b,
                                                 (short)0, c, false, false);
}

// Async global->LDS copy, 16 bytes per lane. vdst VGPR carries the LDS byte
// offset (low 32 bits of a generic LDS pointer == LDS offset), vaddr the
// 32-bit byte offset from the uniform 64-bit SGPR base. Tracked by ASYNCcnt.
static __device__ __forceinline__ void async_ld128(unsigned lds_byte_off,
                                                   const __bf16* base,
                                                   unsigned gbyte_off) {
  asm volatile("global_load_async_to_lds_b128 %0, %1, %2"
               :
               : "v"(lds_byte_off), "v"(gbyte_off), "s"(base)
               : "memory");
}

static __device__ __forceinline__ void wait_async0() {
  asm volatile("s_wait_asynccnt 0x0" ::: "memory");
}

// -------- prep: bf16 hi/lo codebook + per-code squared norms ---------------
__global__ __launch_bounds__(256) void vq_prep(const float* __restrict__ emb,
                                               __bf16* __restrict__ ehi,
                                               __bf16* __restrict__ elo,
                                               float* __restrict__ enorm) {
  const int k = blockIdx.x;
  const int d = threadIdx.x;
  const float x = emb[(size_t)k * DIM + d];
  const __bf16 h = (__bf16)x;
  const float  r = x - (float)h;
  ehi[(size_t)k * DIM + d] = h;
  elo[(size_t)k * DIM + d] = (__bf16)r;

  __shared__ float red[256];
  red[d] = x * x;
  __syncthreads();
  for (int s = 128; s > 0; s >>= 1) {
    if (d < s) red[d] += red[d + s];
    __syncthreads();
  }
  if (d == 0) enorm[k] = red[0];
}

// -------- zero the usage histogram (deterministic across replays) ----------
__global__ void vq_zero(unsigned* __restrict__ usage) {
  const int i = blockIdx.x * blockDim.x + threadIdx.x;
  if (i < KC) usage[i] = 0u;
}

// -------- main: distance GEMM + argmin + gather ----------------------------
__global__ __launch_bounds__(256, 1) void vq_main(
    const float* __restrict__ z_e, const float* __restrict__ emb,
    const __bf16* __restrict__ emb_hi, const __bf16* __restrict__ emb_lo,
    const float* __restrict__ e_norms, unsigned* __restrict__ usage,
    float* __restrict__ out_st, float* __restrict__ out_zq,
    float* __restrict__ out_idx) {
  __shared__ __align__(32) __bf16 lds_hi[2][16 * DIM];   // 2 x 8 KB
  __shared__ __align__(32) __bf16 lds_lo[2][16 * DIM];   // 2 x 8 KB
  __shared__ unsigned idx_lds[QPB];

  const int tid     = threadIdx.x;
  const int wave    = tid >> 5;
  const int lane    = tid & 31;
  const int halfsel = lane >> 4;   // 0: lanes 0-15, 1: lanes 16-31
  const int lidx    = lane & 15;
  const int q0      = blockIdx.x * QPB;
  const int myrow   = q0 + wave * 16 + lidx;

  // ---- A fragments (16 queries x 256 dims) as WMMA A operands, contiguous k order, hi/lo bf16 ---
  // lane<16 : K = c*32 + {0..15} ; lane>=16 : K = c*32 + {16..31}
  v16bf a_hi[8], a_lo[8];
  const float* zrow = z_e + (size_t)myrow * DIM;
#pragma unroll
  for (int c = 0; c < 8; ++c) {
    const int k0 = c * 32 + halfsel * 16;
    const float4 A0 = *(const float4*)(zrow + k0);
    const float4 A1 = *(const float4*)(zrow + k0 + 4);
    const float4 B0 = *(const float4*)(zrow + k0 + 8);
    const float4 B1 = *(const float4*)(zrow + k0 + 12);
    float f[16] = {A0.x, A0.y, A0.z, A0.w, A1.x, A1.y, A1.z, A1.w,
                   B0.x, B0.y, B0.z, B0.w, B1.x, B1.y, B1.z, B1.w};
#pragma unroll
    for (int j = 0; j < 16; ++j) {
      const __bf16 h = (__bf16)f[j];
      a_hi[c][j] = h;
      a_lo[c][j] = (__bf16)(f[j] - (float)h);
    }
  }

  // ---- async double-buffer staging of code tiles --------------------------
  const int r  = tid >> 4;          // code row 0..15 handled by this thread
  const int cs = (tid & 15) * 16;   // 16-element column segment
  const unsigned lbase_hi  = (unsigned)(uintptr_t)&lds_hi[0][0];
  const unsigned lbase_lo  = (unsigned)(uintptr_t)&lds_lo[0][0];
  const unsigned lslot     = (unsigned)(r * DIM + cs) * 2u;  // bytes in buffer
  const unsigned bufstride = (unsigned)(16 * DIM * 2);       // 8192 bytes

#define ISSUE_TILE(T, BUF)                                                    \
  do {                                                                        \
    const unsigned g_  = ((unsigned)((T) * 16 + r) * DIM + cs) * 2u;          \
    const unsigned lh_ = lbase_hi + (unsigned)(BUF) * bufstride + lslot;      \
    const unsigned ll_ = lbase_lo + (unsigned)(BUF) * bufstride + lslot;      \
    async_ld128(lh_,      emb_hi, g_);                                        \
    async_ld128(lh_ + 16, emb_hi, g_ + 16u);                                  \
    async_ld128(ll_,      emb_lo, g_);                                        \
    async_ld128(ll_ + 16, emb_lo, g_ + 16u);                                  \
  } while (0)

  ISSUE_TILE(0, 0);

  // Per-lane running argmin: lane's column slot n=lidx, rows v(+8*halfsel).
  float    bestv[8];
  unsigned bestidx[8];
#pragma unroll
  for (int v = 0; v < 8; ++v) { bestv[v] = 3.0e38f; bestidx[v] = 0u; }

  for (int t = 0; t < NTILE; ++t) {
    const int buf = t & 1;
    wait_async0();        // my slice of tile t has landed in LDS
    __syncthreads();      // => every wave's slice of tile t has landed
    if (t + 1 < NTILE) ISSUE_TILE(t + 1, buf ^ 1);  // overlaps compute below

    const float en = e_norms[t * 16 + lidx];        // L2-resident, hidden

    // Three independent accumulators break the WMMA RAW chain (depth 8, not 24)
    v8f acc_hh = {}, acc_hl = {}, acc_lh = {};
#pragma unroll
    for (int c = 0; c < 8; ++c) {
      // B operand, same k order as A: lane<16 -> col n=lane, K=c*32+j ; lane>=16 -> K=c*32+16+j
      const int off = buf * (16 * DIM) + lidx * DIM + c * 32 + halfsel * 16;
      const v16bf bh = *(const v16bf*)&lds_hi[0][off];
      const v16bf bl = *(const v16bf*)&lds_lo[0][off];
      acc_hh = wmma_bf16(a_hi[c], bh, acc_hh);
      acc_hl = wmma_bf16(a_hi[c], bl, acc_hl);
      acc_lh = wmma_bf16(a_lo[c], bh, acc_lh);
    }

    const unsigned gi = (unsigned)(t * 16 + lidx);
#pragma unroll
    for (int v = 0; v < 8; ++v) {
      const float dot = acc_hh[v] + (acc_hl[v] + acc_lh[v]);
      const float s   = en - 2.0f * dot;  // ||e||^2 - 2 x.e  (||x||^2 const)
      if (s < bestv[v]) { bestv[v] = s; bestidx[v] = gi; }  // strict < => first
    }
  }
#undef ISSUE_TILE

  // ---- argmin reduce across the 16 column slots (within each half-wave) ---
#pragma unroll
  for (int v = 0; v < 8; ++v) {
    float    val = bestv[v];
    unsigned bi  = bestidx[v];
#pragma unroll
    for (int m = 1; m <= 8; m <<= 1) {
      const float    ov = __shfl_xor(val, m, 32);
      const unsigned oi = (unsigned)__shfl_xor((int)bi, m, 32);
      if (ov < val || (ov == val && oi < bi)) { val = ov; bi = oi; }
    }
    if (lidx == 0) {
      const int rl = wave * 16 + v + 8 * halfsel;   // local row 0..127
      idx_lds[rl]       = bi;
      out_idx[q0 + rl]  = (float)bi;
      atomicAdd(&usage[bi], 1u);
    }
  }
  __syncthreads();

  // ---- gather z_q and z_q_st = z_e + (z_q - z_e) (128 rows, 2 thr/row) ----
  {
    const int rr = tid >> 1;   // 0..127
    const int h  = tid & 1;    // half of the 256-dim row
    const unsigned ci = idx_lds[rr];
    const size_t ob = (size_t)(q0 + rr) * DIM + h * 128;
    const float4* src = (const float4*)(emb + (size_t)ci * DIM + h * 128);
    const float4* zs  = (const float4*)(z_e + ob);
    float4* d1 = (float4*)(out_st + ob);
    float4* d2 = (float4*)(out_zq + ob);
#pragma unroll 4
    for (int j = 0; j < 32; ++j) {
      const float4 e4 = src[j];
      const float4 z4 = zs[j];
      float4 st;
      st.x = z4.x + (e4.x - z4.x);
      st.y = z4.y + (e4.y - z4.y);
      st.z = z4.z + (e4.z - z4.z);
      st.w = z4.w + (e4.w - z4.w);
      d1[j] = st;
      d2[j] = e4;
    }
  }
}

// -------- stats: perplexity + dead ratio -----------------------------------
__global__ __launch_bounds__(256) void vq_stats(const unsigned* __restrict__ usage,
                                                float* __restrict__ stats) {
  __shared__ float red[256];
  const int tid = threadIdx.x;

  float lt = 0.f;
  for (int i = tid; i < KC; i += 256) lt += (float)usage[i];
  red[tid] = lt;
  __syncthreads();
  for (int s = 128; s > 0; s >>= 1) {
    if (tid < s) red[tid] += red[tid + s];
    __syncthreads();
  }
  const float total = fmaxf(red[0], 1.0f);
  __syncthreads();

  float le = 0.f, ld = 0.f;
  for (int i = tid; i < KC; i += 256) {
    const float u = (float)usage[i];
    if (u > 0.f) {
      const float p = u / total;
      le += p * logf(p);
    } else {
      ld += 1.f;
    }
  }
  red[tid] = le;
  __syncthreads();
  for (int s = 128; s > 0; s >>= 1) {
    if (tid < s) red[tid] += red[tid + s];
    __syncthreads();
  }
  const float ent = red[0];
  __syncthreads();
  red[tid] = ld;
  __syncthreads();
  for (int s = 128; s > 0; s >>= 1) {
    if (tid < s) red[tid] += red[tid + s];
    __syncthreads();
  }
  if (tid == 0) {
    stats[0] = expf(-ent);              // perplexity
    stats[1] = red[0] / (float)KC;      // dead ratio
  }
}

// ---------------------------------------------------------------------------
extern "C" void kernel_launch(void* const* d_in, const int* in_sizes, int n_in,
                              void* d_out, int out_size, void* d_ws, size_t ws_size,
                              hipStream_t stream) {
  (void)in_sizes; (void)n_in; (void)out_size; (void)ws_size;

  const float* z_e = (const float*)d_in[0];   // [16,2048,256] f32
  const float* emb = (const float*)d_in[1];   // [8192,256]    f32

  // workspace: emb_hi(4MB) | emb_lo(4MB) | e_norms(32KB) | usage(32KB)
  __bf16*  ehi   = (__bf16*)d_ws;
  __bf16*  elo   = ehi + (size_t)KC * DIM;
  float*   enorm = (float*)(elo + (size_t)KC * DIM);
  unsigned* usage = (unsigned*)(enorm + KC);

  // output layout (f32, concatenated in return order)
  float* out   = (float*)d_out;
  float* zq_st = out;                                  // [N,256]
  float* zq    = out + (size_t)NQ * DIM;               // [N,256]
  float* oidx  = out + (size_t)2 * NQ * DIM;           // [N]
  float* stats = oidx + NQ;                            // [2]

  vq_prep <<<KC, 256, 0, stream>>>(emb, ehi, elo, enorm);
  vq_zero <<<(KC + 255) / 256, 256, 0, stream>>>(usage);
  vq_main <<<NQ / QPB, 256, 0, stream>>>(z_e, emb, ehi, elo, enorm, usage,
                                         zq_st, zq, oidx);
  vq_stats<<<1, 256, 0, stream>>>(usage, stats);
}
